// AdaptiveMLP_13932873908505
// MI455X (gfx1250) — compile-verified, matrix operand order checked by reading, not run on hardware
//
#include <hip/hip_runtime.h>
#include <hip/hip_bf16.h>
#include <stdint.h>

// ---------------------------------------------------------------------------
// AdaptiveMLP for MI455X (gfx1250, wave32, WMMA).
// Compute-bound (~5.5e11 FLOPs/path vs ~14us HBM traffic at 23.3 TB/s):
// matmuls run as v_wmma_f32_16x16x32_bf16; zero-gate-weight paths are skipped
// by device-side early exit (stream-serialized after the gate kernel).
// Weights are converted fp32->bf16 AND pre-transposed once per launch so the
// GEMM inner loop stages both operands as linear b128 global->LDS copies
// (async-to-LDS via the gfx1250 builtin, tracked by ASYNCcnt).
// ---------------------------------------------------------------------------

#define H_DIM 2048
#define I_DIM 8192
#define B_DIM 4
#define S_DIM 2048
#define M_DIM (B_DIM * S_DIM) /* 8192 rows */
#define CONF_T 0.6f

typedef __attribute__((ext_vector_type(16))) __bf16 v16bf;
typedef __attribute__((ext_vector_type(8)))  __bf16 v8bf;
typedef __attribute__((ext_vector_type(8)))  float  v8f;
typedef int vec4i __attribute__((vector_size(16)));   // matches builtin param type

#if defined(__has_builtin)
#if __has_builtin(__builtin_amdgcn_global_load_async_to_lds_b128)
#define USE_ASYNC_LDS 1
#endif
#if __has_builtin(__builtin_amdgcn_s_wait_asynccnt)
#define HAVE_WAIT_ASYNC_BUILTIN 1
#endif
#endif
#ifndef USE_ASYNC_LDS
#define USE_ASYNC_LDS 0
#endif

#if USE_ASYNC_LDS
typedef __attribute__((address_space(1))) vec4i* gptr4i;
typedef __attribute__((address_space(3))) vec4i* lptr4i;
static __device__ inline void async_copy16(const void* g, void* l) {
  __builtin_amdgcn_global_load_async_to_lds_b128(
      (gptr4i)(void*)(uintptr_t)g, (lptr4i)l, 0, 0);
}
#endif

static __device__ inline void wait_async0() {
#if defined(HAVE_WAIT_ASYNC_BUILTIN)
  __builtin_amdgcn_s_wait_asynccnt(0);
#else
  asm volatile("s_wait_asynccnt 0" ::: "memory");
#endif
}

static __device__ inline __bf16 f2bf(float f) {
  union { float f; uint32_t u; } c; c.f = f;
  uint32_t u = c.u;
  u += 0x7FFFu + ((u >> 16) & 1u);           // round-to-nearest-even
  uint16_t h = (uint16_t)(u >> 16);
  return __builtin_bit_cast(__bf16, h);
}

static __device__ inline v16bf mk16(v8bf lo, v8bf hi) {
  union { v16bf v; v8bf h[2]; } u;
  u.h[0] = lo; u.h[1] = hi;
  return u.v;
}

// ---------------------------------------------------------------------------
// Zero the output accumulator (paths accumulate into it).
// ---------------------------------------------------------------------------
__global__ void zero_out_k(float* __restrict__ out, int n) {
  int i = blockIdx.x * blockDim.x + threadIdx.x;
  int stride = gridDim.x * blockDim.x;
  for (; i < n; i += stride) out[i] = 0.0f;
}

// ---------------------------------------------------------------------------
// gfeat[b,h] = mean_s x[b,s,h]   (coalesced over h)
// ---------------------------------------------------------------------------
__global__ void reduce_mean_k(const float* __restrict__ x, float* __restrict__ gfeat) {
  int t = blockIdx.x * blockDim.x + threadIdx.x;   // 0..B*H-1
  if (t >= B_DIM * H_DIM) return;
  int b = t >> 11;            // /2048
  int h = t & (H_DIM - 1);
  const float* p = x + (size_t)b * S_DIM * H_DIM + h;
  float s = 0.0f;
  for (int i = 0; i < S_DIM; ++i) s += p[(size_t)i * H_DIM];
  gfeat[t] = s * (1.0f / (float)S_DIM);
}

// ---------------------------------------------------------------------------
// fp32 -> bf16 elementwise (for x)
// ---------------------------------------------------------------------------
__global__ void cvt_bf16_k(const float* __restrict__ src, __bf16* __restrict__ dst, int n) {
  int i = blockIdx.x * blockDim.x + threadIdx.x;
  int stride = gridDim.x * blockDim.x;
  for (; i < n; i += stride) dst[i] = f2bf(src[i]);
}

// ---------------------------------------------------------------------------
// Path-guarded fp32->bf16 weight conversion WITH transpose:
//   W [R][C] row-major fp32  ->  Wt [C][R] row-major bf16.
// LDS-tiled 32x32 so both the global read and the global write are coalesced.
// Moving the transpose here removes all scalar ds_store_b16 traffic from the
// GEMM inner loop (B fragments want K-contiguous data per output column).
// ---------------------------------------------------------------------------
__global__ void cvt_wt_k(const float* __restrict__ W, __bf16* __restrict__ Wt,
                         int R, int C, const float* __restrict__ wbuf, int p) {
  if (wbuf[p] == 0.0f && wbuf[3 + p] == 0.0f &&
      wbuf[6 + p] == 0.0f && wbuf[9 + p] == 0.0f) return;
  __shared__ float t[32][33];
  const int c0 = blockIdx.x * 32;
  const int r0 = blockIdx.y * 32;
  const int tx = threadIdx.x & 31;
  const int ty = threadIdx.x >> 5;    // 0..7 (256 threads)
#pragma unroll
  for (int j = 0; j < 4; ++j)
    t[ty + 8 * j][tx] = W[(size_t)(r0 + ty + 8 * j) * C + c0 + tx];
  __syncthreads();
#pragma unroll
  for (int j = 0; j < 4; ++j)
    Wt[(size_t)(c0 + ty + 8 * j) * R + r0 + tx] = f2bf(t[tx][ty + 8 * j]);
}

// ---------------------------------------------------------------------------
// Gate MLP: relu(gfeat @ gW1 + gb1) @ gW2 + gb2 -> softmax -> confidence mask.
// ---------------------------------------------------------------------------
__global__ void gate_k(const float* __restrict__ gfeat,
                       const float* __restrict__ gW1, const float* __restrict__ gb1,
                       const float* __restrict__ gW2, const float* __restrict__ gb2,
                       float* __restrict__ wout) {
  __shared__ float sg[B_DIM * H_DIM];        // 32 KB
  __shared__ float hid[B_DIM * (H_DIM / 4)]; // 8 KB
  __shared__ float lg[B_DIM * 3];
  int tid = threadIdx.x;                     // 256 threads
  for (int i = tid; i < B_DIM * H_DIM; i += 256) sg[i] = gfeat[i];
  __syncthreads();

  for (int j = tid; j < H_DIM / 4; j += 256) {
    float a0 = 0.f, a1 = 0.f, a2 = 0.f, a3 = 0.f;
    for (int h = 0; h < H_DIM; ++h) {
      float wv = gW1[(size_t)h * (H_DIM / 4) + j];
      a0 += sg[h] * wv;
      a1 += sg[H_DIM + h] * wv;
      a2 += sg[2 * H_DIM + h] * wv;
      a3 += sg[3 * H_DIM + h] * wv;
    }
    float bb = gb1[j];
    hid[j]               = fmaxf(a0 + bb, 0.f);
    hid[(H_DIM/4) + j]   = fmaxf(a1 + bb, 0.f);
    hid[2*(H_DIM/4) + j] = fmaxf(a2 + bb, 0.f);
    hid[3*(H_DIM/4) + j] = fmaxf(a3 + bb, 0.f);
  }
  __syncthreads();

  if (tid < 12) {
    int b = tid / 3, p = tid % 3;
    float s = gb2[p];
    for (int k = 0; k < H_DIM / 4; ++k) s += hid[b * (H_DIM / 4) + k] * gW2[k * 3 + p];
    lg[tid] = s;
  }
  __syncthreads();

  if (tid < B_DIM) {
    float l0 = lg[tid*3], l1 = lg[tid*3+1], l2 = lg[tid*3+2];
    float m = fmaxf(l0, fmaxf(l1, l2));
    float e0 = __expf(l0 - m), e1 = __expf(l1 - m), e2 = __expf(l2 - m);
    float inv = 1.0f / (e0 + e1 + e2);
    float g0 = e0 * inv, g1 = e1 * inv, g2 = e2 * inv;
    int sel = 0; float mx = g0;
    if (g1 > mx) { mx = g1; sel = 1; }
    if (g2 > mx) { mx = g2; sel = 2; }
    bool low = mx < CONF_T;
    wout[tid*3+0] = (low || sel == 0) ? g0 : 0.0f;
    wout[tid*3+1] = (low || sel == 1) ? g1 : 0.0f;
    wout[tid*3+2] = (low || sel == 2) ? g2 : 0.0f;
  }
}

// ---------------------------------------------------------------------------
// Tiled bf16 GEMM: C[128x128] per block, 8 waves, each wave a 32x64 tile of
// 2x4 v_wmma_f32_16x16x32_bf16. Double-buffered LDS, K-step 32.
// A:  [M][K] bf16 row-major.   BT: [N][K] bf16 (pre-transposed weights).
// Both slabs stage as linear b128 async global->LDS copies (ASYNCcnt).
// MODE 0: out = bf16( silu(A@B + bias) )      (out stride N)
// MODE 1: out(f32) += wscale * (A@B + bias)   (out = d_out, stride H_DIM)
// ---------------------------------------------------------------------------
template <int MODE>
__global__ __launch_bounds__(256)
void gemm_k(const __bf16* __restrict__ A, const __bf16* __restrict__ BT,
            const float* __restrict__ bias, void* __restrict__ out,
            int K, int N, const float* __restrict__ wbuf, int p) {
  const int mrow0 = blockIdx.y * 128;
  const int n0    = blockIdx.x * 128;
  const int batch = mrow0 / S_DIM;          // 128 | 2048: tiles never straddle batches
  const float wscale = wbuf[batch * 3 + p];
  if (wscale == 0.0f) return;

  __shared__ __bf16 As[2][128][48];         // 12 KB x2, M x K-slab, padded rows (96B)
  __shared__ __bf16 Bs[2][128][48];         // 12 KB x2, N x K-slab (K-contiguous)

  const int tid  = threadIdx.x;
  const int lane = tid & 31;
  const int wid  = tid >> 5;                // 8 waves
  const int wm   = wid & 3;                 // 4 row-groups of 32
  const int wn   = wid >> 2;                // 2 col-groups of 64

  v8f acc[2][4];
  const v8f vzero = {0.f,0.f,0.f,0.f,0.f,0.f,0.f,0.f};
#pragma unroll
  for (int mi = 0; mi < 2; ++mi)
#pragma unroll
    for (int ni = 0; ni < 4; ++ni) acc[mi][ni] = vzero;

  auto stage = [&](int buf, int k0) {
    // A slab: 128 rows x 32 k = 512 x 16B chunks, 2 per thread
#pragma unroll
    for (int j = 0; j < 2; ++j) {
      int c   = tid + 256 * j;
      int row = c >> 2;
      int kc  = (c & 3) * 8;
      const __bf16* ga = A + (size_t)(mrow0 + row) * K + k0 + kc;
      __bf16* la = &As[buf][row][kc];
#if USE_ASYNC_LDS
      async_copy16(ga, la);
#else
      *(v8bf*)la = *(const v8bf*)ga;
#endif
    }
    // B slab: 128 n-rows x 32 k = 512 x 16B chunks, 2 per thread (K-contiguous)
#pragma unroll
    for (int j = 0; j < 2; ++j) {
      int c  = tid + 256 * j;
      int n  = c >> 2;
      int kc = (c & 3) * 8;
      const __bf16* gb = BT + (size_t)(n0 + n) * K + k0 + kc;
      if (j == 0) __builtin_prefetch(gb + 64, 0, 1);   // global_prefetch_b8, 2 steps ahead
      __bf16* lb = &Bs[buf][n][kc];
#if USE_ASYNC_LDS
      async_copy16(gb, lb);
#else
      *(v8bf*)lb = *(const v8bf*)gb;
#endif
    }
  };

  stage(0, 0);
#if USE_ASYNC_LDS
  wait_async0();
#endif
  __syncthreads();

  const int nK = K / 32;
  const int khalf = (lane >> 4) * 8;        // ISA A/B striping: half-wave K split
  const int mrow  = wm * 32 + (lane & 15);
  const int ncol  = wn * 64 + (lane & 15);

  for (int ks = 0; ks < nK; ++ks) {
    const int cur = ks & 1;
    if (ks + 1 < nK) stage(cur ^ 1, (ks + 1) * 32);

    v16bf af[2];
#pragma unroll
    for (int mi = 0; mi < 2; ++mi) {
      const __bf16* pa = &As[cur][mrow + mi * 16][khalf];
      af[mi] = mk16(*(const v8bf*)pa, *(const v8bf*)(pa + 16));
    }
    v16bf bfr[4];
#pragma unroll
    for (int ni = 0; ni < 4; ++ni) {
      const __bf16* pb = &Bs[cur][ncol + ni * 16][khalf];
      bfr[ni] = mk16(*(const v8bf*)pb, *(const v8bf*)(pb + 16));
    }
#pragma unroll
    for (int mi = 0; mi < 2; ++mi)
#pragma unroll
      for (int ni = 0; ni < 4; ++ni)
        acc[mi][ni] = __builtin_amdgcn_wmma_f32_16x16x32_bf16(
            false, af[mi], false, bfr[ni], (short)0, acc[mi][ni], false, false);

#if USE_ASYNC_LDS
    wait_async0();                          // LDS writes of next buffer landed
#endif
    __syncthreads();
  }

  // Epilogue. C layout (ISA 7.12.2): vgpr r, lane L -> M = r + 8*(L>=16), N = L%16.
  const int rbase = ((lane >> 4) << 3);
#pragma unroll
  for (int mi = 0; mi < 2; ++mi) {
#pragma unroll
    for (int ni = 0; ni < 4; ++ni) {
      int colg = n0 + wn * 64 + ni * 16 + (lane & 15);
      float bv = bias[colg];
#pragma unroll
      for (int r = 0; r < 8; ++r) {
        int rowg = mrow0 + wm * 32 + mi * 16 + rbase + r;
        float v = acc[mi][ni][r] + bv;
        if (MODE == 0) {
          float s = v * (1.0f / (1.0f + __expf(-v)));   // SiLU
          ((__bf16*)out)[(size_t)rowg * N + colg] = f2bf(s);
        } else {
          float* o = (float*)out + (size_t)rowg * H_DIM + colg;
          *o += wscale * v;    // paths stream-serialized; no cross-kernel race
        }
      }
    }
  }
}

// ---------------------------------------------------------------------------
// Host launch. Workspace layout (assumes ws_size >= ~225 MB):
//   [0,32K)            gfeat (B*H f32)
//   [32K,64K)          gate weights w[B][3] f32
//   [64K, +32MB)       x in bf16
//   [+32MB, +96MB)     active path weights bf16, TRANSPOSED (W1t then W2t)
//   [+96MB, +224MB)    intermediate h in bf16 (M x I)
// ---------------------------------------------------------------------------
extern "C" void kernel_launch(void* const* d_in, const int* in_sizes, int n_in,
                              void* d_out, int out_size, void* d_ws, size_t ws_size,
                              hipStream_t stream) {
  const float* x   = (const float*)d_in[0];
  const float* gW1 = (const float*)d_in[1];
  const float* gb1 = (const float*)d_in[2];
  const float* gW2 = (const float*)d_in[3];
  const float* gb2 = (const float*)d_in[4];
  const float* fW1 = (const float*)d_in[5];
  const float* fb1 = (const float*)d_in[6];
  const float* fW2 = (const float*)d_in[7];
  const float* fb2 = (const float*)d_in[8];
  const float* cW1 = (const float*)d_in[9];
  const float* cb1 = (const float*)d_in[10];
  const float* cW2 = (const float*)d_in[11];
  const float* cb2 = (const float*)d_in[12];
  const float* sW1 = (const float*)d_in[13];
  const float* sb1 = (const float*)d_in[14];
  const float* sW2 = (const float*)d_in[15];
  const float* sb2 = (const float*)d_in[16];

  char* ws = (char*)d_ws;
  float*  gfeat = (float*)ws;
  float*  wbuf  = (float*)(ws + 32 * 1024);
  __bf16* xb    = (__bf16*)(ws + 64 * 1024);
  __bf16* wb    = (__bf16*)(ws + 64 * 1024 + (size_t)32 * 1024 * 1024);
  __bf16* hbuf  = (__bf16*)(ws + 64 * 1024 + (size_t)96 * 1024 * 1024);

  zero_out_k<<<1024, 256, 0, stream>>>((float*)d_out, M_DIM * H_DIM);
  cvt_bf16_k<<<2048, 256, 0, stream>>>(x, xb, M_DIM * H_DIM);
  reduce_mean_k<<<(B_DIM * H_DIM) / 256, 256, 0, stream>>>(x, gfeat);
  gate_k<<<1, 256, 0, stream>>>(gfeat, gW1, gb1, gW2, gb2, wbuf);

  struct Path { const float *W1, *b1, *W2, *b2; int N1; };
  Path paths[3] = {
      {fW1, fb1, fW2, fb2, I_DIM},
      {cW1, cb1, cW2, cb2, I_DIM / 2},
      {sW1, sb1, sW2, sb2, I_DIM},
  };

  for (int p = 0; p < 3; ++p) {
    const int N1 = paths[p].N1;
    __bf16* W1t = wb;                          // [N1][H]
    __bf16* W2t = wb + (size_t)H_DIM * N1;     // [H][N1]

    // W1 [H][N1] -> W1t [N1][H] ; W2 [N1][H] -> W2t [H][N1]   (path-guarded)
    cvt_wt_k<<<dim3(N1 / 32, H_DIM / 32), 256, 0, stream>>>(
        paths[p].W1, W1t, H_DIM, N1, wbuf, p);
    cvt_wt_k<<<dim3(H_DIM / 32, N1 / 32), 256, 0, stream>>>(
        paths[p].W2, W2t, N1, H_DIM, wbuf, p);

    dim3 g1(N1 / 128, M_DIM / 128);
    gemm_k<0><<<g1, 256, 0, stream>>>(xb, W1t, paths[p].b1, (void*)hbuf,
                                      H_DIM, N1, wbuf, p);
    dim3 g2(H_DIM / 128, M_DIM / 128);
    gemm_k<1><<<g2, 256, 0, stream>>>(hbuf, W2t, paths[p].b2, d_out,
                                      N1, H_DIM, wbuf, p);
  }
}
